// CRFClassifier_81269371175415
// MI455X (gfx1250) — matrix operand order repaired and checked by reading, not compiled, not run on hardware
//
#include <hip/hip_runtime.h>
#include <hip/hip_bf16.h>
#include <math.h>

typedef __attribute__((ext_vector_type(16))) _Float16 v16h;
typedef __attribute__((ext_vector_type(8)))  _Float16 v8h;
typedef __attribute__((ext_vector_type(8)))  float    v8f;

#define B_SZ 128
#define T_SZ 512
#define L_SZ 256
#define NEGV (-100.0f)
#define LN2   0.6931471805599453f
#define LOG2E 1.4426950408889634f

// ---------------------------------------------------------------------------
// CRF forward scan. One workgroup per 16 batch rows; 16 waves (wave32).
// Wave w owns output j-block w (cols w*16..w*16+15) and keeps the 8
// B-fragments of E = exp(transit) resident in VGPRs for the whole scan.
// Per time step: preload all 8 A-fragments (pipelined ds_load_b128), 8
// chained v_wmma_f32_16x16x32_f16 (K=256), then wave r finalizes batch
// row r. x[t+1] is prefetched one step ahead to hide HBM latency.
// fv state lives in the finalizing wave's registers.
// ---------------------------------------------------------------------------
__launch_bounds__(512, 1)
__global__ void crf_scan_kernel(const float* __restrict__ x,
                                const float* __restrict__ transit,
                                const int*   __restrict__ lengths,
                                float*       __restrict__ alpha_ws)
{
    __shared__ __align__(16) _Float16 pbuf[16][264];   // p = exp(fv - max), f16 A-matrix
    __shared__ __align__(16) float    nxtbuf[16][268]; // GEMM result staging

    const int tid    = threadIdx.x;
    const int wave   = tid >> 5;            // 0..15 : j-block AND finalize row
    const int lane   = tid & 31;
    const int bblock = blockIdx.x;          // 0..7
    const int row    = wave;                // batch row this wave finalizes
    const int bidx   = bblock * 16 + row;   // global batch index
    const int len_s  = __builtin_amdgcn_readfirstlane(lengths[bidx]);

    // ---- persistent B fragments: B[k][n] = E[j][k] = exp(transit[j][k]) ----
    // lane l: n = l&15  -> j = wave*16 + n ;  k-sub-block = (l>>4)*16, linear in i
    const int jcol  = wave * 16 + (lane & 15);
    const int khalf = (lane >> 4) * 16;
    v16h bfrag[8];
#pragma unroll
    for (int kb = 0; kb < 8; ++kb) {
        const float* tr = transit + (size_t)jcol * L_SZ + kb * 32 + khalf;
#pragma unroll
        for (int i = 0; i < 16; ++i)
            bfrag[kb][i] = (_Float16)__expf(tr[i]);
    }

    // ---- init state: fv[:,0] = 0, else -100 ; p = exp(fv - 0) ----
    for (int idx = tid; idx < 16 * 256; idx += 512) {
        int m = idx >> 8, k = idx & 255;
        pbuf[m][k] = (k == 0) ? (_Float16)1.0f : (_Float16)0.0f;
    }
    float fv[8];                            // fv row `row`, cols lane*8..lane*8+7
#pragma unroll
    for (int i = 0; i < 8; ++i)
        fv[i] = (lane == 0 && i == 0) ? 0.0f : NEGV;
    float m_prev = 0.0f;                    // running row max (per-wave register)

    // x prefetch registers for this wave's row (cols lane*8..+7)
    const float* xbase = x + ((size_t)bidx * T_SZ) * L_SZ + lane * 8;
    float xv[8];
    {
        const float4 a0 = *(const float4*)(xbase + 0);
        const float4 a1 = *(const float4*)(xbase + 4);
        xv[0]=a0.x; xv[1]=a0.y; xv[2]=a0.z; xv[3]=a0.w;
        xv[4]=a1.x; xv[5]=a1.y; xv[6]=a1.z; xv[7]=a1.w;
    }
    __syncthreads();

    const int am = lane & 15;               // A-matrix row this lane loads
    const int ak = (lane >> 4) * 16;        // A-matrix k sub-offset
    const int cn = wave * 16 + (lane & 15); // C-fragment column
    const int mbase = (lane >> 4) * 8;      // C-fragment row base

    for (int t = 0; t < T_SZ; ++t) {
        // ---- prefetch x for step t+1 (overlaps GEMM + barriers) ----
        float xn[8];
        {
            const int tn = (t + 1 < T_SZ) ? (t + 1) : t;
            const float* xp = xbase + (size_t)tn * L_SZ;
            const float4 a0 = *(const float4*)(xp + 0);
            const float4 a1 = *(const float4*)(xp + 4);
            xn[0]=a0.x; xn[1]=a0.y; xn[2]=a0.z; xn[3]=a0.w;
            xn[4]=a1.x; xn[5]=a1.y; xn[6]=a1.z; xn[7]=a1.w;
        }

        // ---- load all 8 A fragments up front (pipelined DS reads) ----
        v16h afrag[8];
#pragma unroll
        for (int kb = 0; kb < 8; ++kb) {
            v8h lo = *(const v8h*)&pbuf[am][kb * 32 + ak];
            v8h hi = *(const v8h*)&pbuf[am][kb * 32 + ak + 8];
#pragma unroll
            for (int i = 0; i < 8; ++i) { afrag[kb][i] = lo[i]; afrag[kb][i + 8] = hi[i]; }
        }

        // ---- GEMM: acc[m][j] = sum_k p[m][k] * E[j][k] ----
        v8f acc = {};
#pragma unroll
        for (int kb = 0; kb < 8; ++kb)
            acc = __builtin_amdgcn_wmma_f32_16x16x32_f16(
                      false, afrag[kb], false, bfrag[kb], (short)0, acc, false, false);

        // ---- scatter C fragment: VGPR r -> row mbase+r, col cn ----
#pragma unroll
        for (int r2 = 0; r2 < 8; ++r2)
            nxtbuf[mbase + r2][cn] = acc[r2];
        __syncthreads();

        // ---- finalize row `row` (scalar-uniform mask branch) ----
        if (t < len_s) {
            float vals[8];
            float vmax = -INFINITY;
#pragma unroll
            for (int i = 0; i < 8; ++i) {
                // log(acc) = log2(acc)*ln2 ; acc >= ~0.9, no denorm guard needed
                float lg = __builtin_amdgcn_logf(nxtbuf[row][lane * 8 + i]);
                float v  = fmaf(lg, LN2, m_prev + xv[i]);
                vals[i] = v;
                vmax = fmaxf(vmax, v);
            }
#pragma unroll
            for (int off = 16; off > 0; off >>= 1)
                vmax = fmaxf(vmax, __shfl_xor(vmax, off, 32));
            m_prev = vmax;
            v8h ph;
#pragma unroll
            for (int i = 0; i < 8; ++i) {
                fv[i] = vals[i];
                ph[i] = (_Float16)__builtin_amdgcn_exp2f((vals[i] - vmax) * LOG2E);
            }
            *(v8h*)&pbuf[row][lane * 8] = ph;   // 16B contiguous store
        }
#pragma unroll
        for (int i = 0; i < 8; ++i) xv[i] = xn[i];
        __syncthreads();
    }

    // ---- alpha[b] = logsumexp_k( fv[k] + transit[L-1][k] ) ----
    {
        const float* trL = transit + (size_t)(L_SZ - 1) * L_SZ + lane * 8;
        float vals[8];
        float vmax = -INFINITY;
#pragma unroll
        for (int i = 0; i < 8; ++i) {
            float v = fv[i] + trL[i];
            vals[i] = v;
            vmax = fmaxf(vmax, v);
        }
#pragma unroll
        for (int off = 16; off > 0; off >>= 1)
            vmax = fmaxf(vmax, __shfl_xor(vmax, off, 32));
        float s = 0.0f;
#pragma unroll
        for (int i = 0; i < 8; ++i)
            s += __builtin_amdgcn_exp2f((vals[i] - vmax) * LOG2E);
#pragma unroll
        for (int off = 16; off > 0; off >>= 1)
            s += __shfl_xor(s, off, 32);
        if (lane == 0)
            alpha_ws[bidx] = fmaf(__builtin_amdgcn_logf(s), LN2, vmax);
    }
}

// ---------------------------------------------------------------------------
// Gold (given-sequence) score: one block per batch element.
// ---------------------------------------------------------------------------
__global__ void gold_kernel(const float* __restrict__ x,
                            const float* __restrict__ transit,
                            const int*   __restrict__ y,
                            const int*   __restrict__ lengths,
                            float*       __restrict__ gold_ws)
{
    __shared__ float red[256];
    const int b    = blockIdx.x;
    const int tid  = threadIdx.x;
    const int lenb = lengths[b];
    float s = 0.0f;
    // emission score
    for (int t = tid; t < T_SZ; t += 256) {
        if (t < lenb)
            s += x[((size_t)b * T_SZ + t) * L_SZ + y[b * T_SZ + t]];
    }
    // transition score over padded sequence [0, y..., L-1]
    for (int t = tid; t <= T_SZ; t += 256) {
        if (t < lenb + 1) {
            int from = (t == 0)    ? 0           : y[b * T_SZ + t - 1];
            int to   = (t == T_SZ) ? (L_SZ - 1)  : y[b * T_SZ + t];
            s += transit[to * L_SZ + from];
        }
    }
    red[tid] = s;
    __syncthreads();
    for (int off = 128; off > 0; off >>= 1) {
        if (tid < off) red[tid] += red[tid + off];
        __syncthreads();
    }
    if (tid == 0) gold_ws[b] = red[0];
}

// ---------------------------------------------------------------------------
// mean(alpha - gold)
// ---------------------------------------------------------------------------
__global__ void mean_kernel(const float* __restrict__ alpha_ws,
                            const float* __restrict__ gold_ws,
                            float*       __restrict__ out)
{
    __shared__ float red[128];
    const int tid = threadIdx.x;
    red[tid] = alpha_ws[tid] - gold_ws[tid];
    __syncthreads();
    for (int off = 64; off > 0; off >>= 1) {
        if (tid < off) red[tid] += red[tid + off];
        __syncthreads();
    }
    if (tid == 0) out[0] = red[0] * (1.0f / (float)B_SZ);
}

extern "C" void kernel_launch(void* const* d_in, const int* in_sizes, int n_in,
                              void* d_out, int out_size, void* d_ws, size_t ws_size,
                              hipStream_t stream)
{
    const float* x       = (const float*)d_in[0];   // [128][512][256] f32
    const float* transit = (const float*)d_in[1];   // [256][256] f32
    const int*   y       = (const int*)  d_in[2];   // [128][512] i32
    const int*   lengths = (const int*)  d_in[3];   // [128] i32

    float* alpha_ws = (float*)d_ws;          // 128 floats
    float* gold_ws  = alpha_ws + B_SZ;       // 128 floats

    gold_kernel<<<B_SZ, 256, 0, stream>>>(x, transit, y, lengths, gold_ws);
    crf_scan_kernel<<<B_SZ / 16, 512, 0, stream>>>(x, transit, lengths, alpha_ws);
    mean_kernel<<<1, 128, 0, stream>>>(alpha_ws, gold_ws, (float*)d_out);
}
